// EIRnn_61658550502004
// MI455X (gfx1250) — matrix operand order repaired and checked by reading, not compiled, not run on hardware
//
#include <hip/hip_runtime.h>
#include <stdint.h>

// Problem constants (from reference): V=32000, E=512, N=2048, S=1024
#define E_ 512
#define N_ 2048
#define S_ 1024

typedef __attribute__((ext_vector_type(16))) __bf16        v16bf;
typedef __attribute__((ext_vector_type(8)))  float         v8f;
typedef __attribute__((ext_vector_type(4)))  unsigned int  v4u;

union ABu { v4u u[2]; v16bf v; };   // 32 bytes = 16 bf16 (trivial types only)

__device__ __forceinline__ unsigned short f32_to_bf16_rne(float f) {
    unsigned u = __float_as_uint(f);
    u += 0x7FFFu + ((u >> 16) & 1u);     // round-to-nearest-even
    return (unsigned short)(u >> 16);
}

// ---------------------------------------------------------------------------
// Kernel 1: gather embedding rows + relu(w_in), convert both to bf16.
// ---------------------------------------------------------------------------
__global__ void k_prep(const int* __restrict__ tokens,
                       const float* __restrict__ emb,
                       const float* __restrict__ w_in,
                       unsigned short* __restrict__ Abf,   // S x E
                       unsigned short* __restrict__ Bbf) { // N x E
    int idx = blockIdx.x * blockDim.x + threadIdx.x;
    const int nA = S_ * E_;
    const int nB = N_ * E_;
    if (idx < nA) {
        int t = idx >> 9;            // / E_
        int e = idx & (E_ - 1);
        float v = emb[(size_t)tokens[t] * E_ + e];
        Abf[idx] = f32_to_bf16_rne(v);
    } else if (idx < nA + nB) {
        int i2 = idx - nA;
        float v = w_in[i2];
        Bbf[i2] = f32_to_bf16_rne(v > 0.f ? v : 0.f);
    }
}

// ---------------------------------------------------------------------------
// Kernel 2: inp_w(S,N) = A(S,E) @ B(N,E)^T  via v_wmma_f32_16x16x32_bf16.
// One wave per 16x16 C tile, K-loop over E in steps of 32.
// A 16x32 bf16 layout: lanes 0-15 -> M=lane,  K 0..7 (v0..3) & 16..23 (v4..7)
//                      lanes16-31 -> M=lane-16,K 8..15 & 24..31
// B 32x16 layout:      lanes 0-15 -> N=lane,  K 0..7 & 16..23; lanes16-31 -> K 8..15 & 24..31
// C layout: VGPR r holds M=r (lanes 0-15) / M=r+8 (lanes 16-31), N=lane%16.
// ---------------------------------------------------------------------------
__global__ void k_gemm(const unsigned short* __restrict__ Abf,
                       const unsigned short* __restrict__ Bbf,
                       float* __restrict__ C) {
    int wid  = blockIdx.x * (blockDim.x >> 5) + (threadIdx.x >> 5);
    int lane = threadIdx.x & 31;
    int tm = wid & 63;            // 64 M-tiles
    int tn = wid >> 6;            // 128 N-tiles
    int m0 = tm << 4, n0 = tn << 4;
    int l16 = lane & 15;
    int hi  = lane >> 4;          // 0: low half-wave, 1: high half-wave

    const v4u* Arow = reinterpret_cast<const v4u*>(Abf + (size_t)(m0 + l16) * E_);
    const v4u* Brow = reinterpret_cast<const v4u*>(Bbf + (size_t)(n0 + l16) * E_);

    v8f acc = {0.f, 0.f, 0.f, 0.f, 0.f, 0.f, 0.f, 0.f};
    for (int k0 = 0; k0 < E_; k0 += 32) {
        ABu a, b;
        int c8 = k0 >> 3;                 // offset in 8-bf16 (16B) units
        a.u[0] = Arow[c8 + hi];           // K = k0 + hi*8 .. +7
        a.u[1] = Arow[c8 + 2 + hi];       // K = k0 + 16 + hi*8 .. +7
        b.u[0] = Brow[c8 + hi];           // K = k0 + hi*8 .. +7
        b.u[1] = Brow[c8 + 2 + hi];
        acc = __builtin_amdgcn_wmma_f32_16x16x32_bf16(false, a.v, false, b.v,
                                                      (short)0, acc, false, false);
    }
#pragma unroll
    for (int r = 0; r < 8; ++r) {
        int mm = m0 + r + (hi << 3);
        C[(size_t)mm * N_ + n0 + l16] = acc[r];
    }
}

// ---------------------------------------------------------------------------
// Kernel 3: persistent-grid EI-RNN scan. 64 WGs x 256 thr; WG g owns columns
// [32g, 32g+32). w_rec slab (relu * EI-sign applied) lives in LDS (256 KB).
// Per step: stage relu(state) (8KB) in LDS, 8-way split-row matvec, LDS
// reduce, owners update their 32 state entries, grid barrier (atomic ctr).
// Ping-pong state buffers; one barrier per step is race-free.
// ---------------------------------------------------------------------------
__global__ void __launch_bounds__(256, 1)
k_rnn(const float* __restrict__ w_rec_plus,
      const float* __restrict__ inp_w,     // S x N
      float* __restrict__ st,              // two N_ buffers (ping-pong)
      unsigned* __restrict__ ctr,
      int nblocks) {
    extern __shared__ float lds[];
    float* wslab = lds;                    // N_ x 32
    float* s_sh  = lds + N_ * 32;          // N_
    float* red   = s_sh + N_;              // 256

    const int tid = threadIdx.x;
    const int j   = tid & 31;
    const int grp = tid >> 5;              // 0..7 (wave id)
    const int c0  = blockIdx.x << 5;
    const int col = c0 + j;
    const float sgn = ((float)col < 0.8f * (float)N_) ? 1.f : -1.f;

    // One-time slab load: w_rec[:, col] = relu(w_rec_plus) * sign(col)
    for (int i = grp; i < N_; i += 8) {
        float w = w_rec_plus[(size_t)i * N_ + col];
        wslab[i * 32 + j] = (w > 0.f ? w : 0.f) * sgn;
    }
    __syncthreads();

    const float alpha  = 0.001f;           // dt/tau
    const float kNoise = 1.41421356e-6f;   // sqrt(2*alpha*var_rec)*1e-3

    for (int t = 0; t < S_; ++t) {
        const float* sprev = st + (size_t)(t & 1) * N_;
        float*       snext = st + (size_t)((t + 1) & 1) * N_;

        for (int k = tid; k < N_; k += 256) {
            float v = sprev[k];
            s_sh[k] = v > 0.f ? v : 0.f;
        }
        __syncthreads();

        float acc = 0.f;
        const int i0 = grp << 8;           // 256 rows per wave
#pragma unroll 8
        for (int ii = 0; ii < 256; ++ii) {
            int i = i0 + ii;
            acc = fmaf(s_sh[i], wslab[i * 32 + j], acc);
        }
        red[tid] = acc;
        __syncthreads();

        if (tid < 32) {
            float h = red[j] + red[32 + j] + red[64 + j] + red[96 + j]
                    + red[128 + j] + red[160 + j] + red[192 + j] + red[224 + j];
            float x  = inp_w[(size_t)t * N_ + col];
            float sp = sprev[col];
            snext[col] = (1.f - alpha) * sp + alpha * (h + x) + kNoise;
            if (t + 1 < S_)                 // warm next step's input slice
                __builtin_prefetch(&inp_w[(size_t)(t + 1) * N_ + col], 0, 0);
        }
        __threadfence();
        __syncthreads();
        if (tid == 0) {                    // grid barrier: monotonic counter
            atomicAdd(ctr, 1u);
            unsigned target = (unsigned)nblocks * (unsigned)(t + 1);
            volatile unsigned* vc = (volatile unsigned*)ctr;
            while (*vc < target) { __builtin_amdgcn_s_sleep(2); }
            __threadfence();
        }
        __syncthreads();
    }
}

// ---------------------------------------------------------------------------
// Kernel 4: last_out partials = relu(state) @ relu(w_out), split-K over j
// (8 deterministic partial buffers, no float atomics -> replay-stable).
// ---------------------------------------------------------------------------
__global__ void k_out_partial(const float* __restrict__ stateF,   // N_
                              const float* __restrict__ w_out,    // N x S
                              float* __restrict__ partial) {      // 8 x S
    __shared__ float stc[256];
    int tid = threadIdx.x;
    int s = blockIdx.x * 256 + tid;
    int jbase = blockIdx.y * 256;
    float v = stateF[jbase + tid];
    stc[tid] = v > 0.f ? v : 0.f;
    __syncthreads();
    float acc = 0.f;
    for (int jj = 0; jj < 256; ++jj) {
        float w = w_out[(size_t)(jbase + jj) * S_ + s];
        acc = fmaf(stc[jj], w > 0.f ? w : 0.f, acc);
    }
    partial[blockIdx.y * S_ + s] = acc;
}

// ---------------------------------------------------------------------------
// Kernel 5: reduce partials, 2-wide MLP, stable softmax.
// ---------------------------------------------------------------------------
__global__ void k_finalize(const float* __restrict__ partial,
                           const float* __restrict__ mlp_w,   // 2 x S
                           const float* __restrict__ mlp_b,   // 2
                           float* __restrict__ out) {         // 2
    __shared__ float r0[256], r1[256];
    int tid = threadIdx.x;
    float l0 = 0.f, l1 = 0.f;
    for (int s = tid; s < S_; s += 256) {
        float v = 0.f;
#pragma unroll
        for (int p = 0; p < 8; ++p) v += partial[p * S_ + s];
        l0 = fmaf(mlp_w[s], v, l0);
        l1 = fmaf(mlp_w[S_ + s], v, l1);
    }
    r0[tid] = l0; r1[tid] = l1;
    __syncthreads();
    for (int off = 128; off > 0; off >>= 1) {
        if (tid < off) { r0[tid] += r0[tid + off]; r1[tid] += r1[tid + off]; }
        __syncthreads();
    }
    if (tid == 0) {
        float a = r0[0] + mlp_b[0];
        float b = r1[0] + mlp_b[1];
        float m = a > b ? a : b;
        float e0 = __expf(a - m), e1 = __expf(b - m);
        float inv = 1.f / (e0 + e1);
        out[0] = e0 * inv;
        out[1] = e1 * inv;
    }
}

// ---------------------------------------------------------------------------
extern "C" void kernel_launch(void* const* d_in, const int* in_sizes, int n_in,
                              void* d_out, int out_size, void* d_ws, size_t ws_size,
                              hipStream_t stream) {
    (void)in_sizes; (void)n_in; (void)out_size; (void)ws_size;
    const int*   tokens    = (const int*)d_in[0];
    const float* embedding = (const float*)d_in[1];
    const float* w_in      = (const float*)d_in[2];
    const float* w_rec     = (const float*)d_in[3];
    const float* w_out     = (const float*)d_in[4];
    const float* mlp_w     = (const float*)d_in[5];
    const float* mlp_b     = (const float*)d_in[6];
    const float* state0    = (const float*)d_in[7];

    // Workspace layout (~11.6 MB)
    char* ws = (char*)d_ws;
    float*          inp_w   = (float*)ws;          ws += (size_t)S_ * N_ * 4;
    unsigned short* Abf     = (unsigned short*)ws; ws += (size_t)S_ * E_ * 2;
    unsigned short* Bbf     = (unsigned short*)ws; ws += (size_t)N_ * E_ * 2;
    float*          st      = (float*)ws;          ws += (size_t)2 * N_ * 4;
    float*          partial = (float*)ws;          ws += (size_t)8 * S_ * 4;
    unsigned*       ctr     = (unsigned*)ws;       ws += 128;

    // state ping-pong buffer 0 <- state0; barrier counter <- 0 (every call,
    // deterministic, capture-safe)
    hipMemcpyAsync(st, state0, N_ * sizeof(float), hipMemcpyDeviceToDevice, stream);
    hipMemsetAsync(ctr, 0, sizeof(unsigned), stream);

    k_prep<<<(S_ * E_ + N_ * E_ + 255) / 256, 256, 0, stream>>>(
        tokens, embedding, w_in, Abf, Bbf);

    // 8192 C-tiles, 8 waves/block -> 1024 blocks
    k_gemm<<<(S_ / 16) * (N_ / 16) / 8, 256, 0, stream>>>(Abf, Bbf, inp_w);

    const int rnnBlocks = N_ / 32;                       // 64 persistent WGs
    const size_t ldsBytes = (size_t)(N_ * 32 + N_ + 256) * sizeof(float); // 265 KB
    hipFuncSetAttribute((const void*)k_rnn,
                        hipFuncAttributeMaxDynamicSharedMemorySize, (int)ldsBytes);
    k_rnn<<<rnnBlocks, 256, ldsBytes, stream>>>(w_rec, inp_w, st, ctr, rnnBlocks);

    // final state lives in buffer 0 ((1023+1)&1 == 0)
    k_out_partial<<<dim3(S_ / 256, N_ / 256), 256, 0, stream>>>(st, w_out, partial);
    k_finalize<<<1, 256, 0, stream>>>(partial, mlp_w, mlp_b, (float*)d_out);
}
